// ResContextBlock_28939489640665
// MI455X (gfx1250) — matrix operand order repaired and checked by reading, not compiled, not run on hardware
//
#include <hip/hip_runtime.h>

typedef __attribute__((ext_vector_type(16))) _Float16 v16h;
typedef __attribute__((ext_vector_type(8)))  float    v8f;

constexpr int DDIM = 16;
constexpr int HDIM = 128;
constexpr int WDIM = 128;
constexpr int NVOX = DDIM * HDIM * WDIM;   // 262144 voxels
constexpr int CIN  = 64;

// Workspace byte offsets (all multiples of 128 -> vector-load aligned)
constexpr size_t WP1A_OFF = 0;                   // 9*64*64  f16 = 73728 B
constexpr size_t WP1B_OFF = 73728;               // 9*128*64 f16 = 147456 B
constexpr size_t WP2A_OFF = 221184;              // 73728 B
constexpr size_t WP2B_OFF = 294912;              // 147456 B
constexpr size_t MASK_OFF = 442368;              // NVOX f32 = 1 MiB
constexpr size_t XT_OFF   = 1490944;             // NVOX*64 f16 = 32 MiB
constexpr size_t S16_OFF  = 35045376;            // NVOX*64 f16 = 32 MiB (reused)
constexpr size_t TRAW_OFF = 68599808;            // NVOX*128 f32 = 128 MiB (shortcut)
// total ~193 MiB

// ---------------------------------------------------------------------------
// Repack fp32 weights into the CDNA5 16-bit A-matrix operand layout:
// dst[tap][mtile][kchunk][lane][h].  Contiguous k order, same as the B loads (16-bit A, 16x32):
// lanes 0-15 hold M=lane; half h -> klocal = (lane/16)*16 + h.
// ---------------------------------------------------------------------------
__global__ void k_pack_weights(const float* __restrict__ W,
                               _Float16* __restrict__ dst,
                               int Cout) {
  int idx = blockIdx.x * blockDim.x + threadIdx.x;
  int total = 9 * Cout * CIN;
  if (idx >= total) return;
  int h    = idx & 15;
  int lane = (idx >> 4) & 31;
  int rest = idx >> 9;
  int kc   = rest & 1;
  rest   >>= 1;
  int MT   = Cout >> 4;
  int mt   = rest % MT;
  int tap  = rest / MT;
  int m      = mt * 16 + (lane & 15);
  int klocal = ((lane >> 4) << 4) + h;
  int k      = kc * 32 + klocal;
  dst[idx] = (_Float16)W[(m * CIN + k) * 9 + tap];
}

// ---------------------------------------------------------------------------
// NCDHW f32 -> NDHWC f16; derive the active-voxel mask from pre-masked x.
// ---------------------------------------------------------------------------
__global__ void k_transform(const float* __restrict__ x,
                            _Float16* __restrict__ xT,
                            float* __restrict__ maskf) {
  int v = blockIdx.x * blockDim.x + threadIdx.x;
  if (v >= NVOX) return;
  int active = 0;
  #pragma unroll 8
  for (int c = 0; c < CIN; ++c) {
    float val = x[(size_t)c * NVOX + v];   // coalesced across lanes per c
    active |= (val != 0.0f);
    xT[(size_t)v * CIN + c] = (_Float16)val;
  }
  maskf[v] = active ? 1.0f : 0.0f;
}

// ---------------------------------------------------------------------------
// Implicit-GEMM submanifold conv via v_wmma_f32_16x16x32_f16, with the
// input halo staged into LDS (zero-padded there -> divergence-free MMA loop)
// and the GroupNorm epilogue fused into the conv (GN groups of 2/4 channels
// are thread-local in the WMMA D layout).
//
// TYPE: 9-tap footprint  0:(3,3,1) over D,H   1:(1,3,3) over H,W   2:(3,1,3) over D,W
// EPI:  0 = GN(groups of 2)+LeakyReLU+mask -> f16 NDHWC   (mid conv, COUT=64)
//       1 = GN(groups of 4)+mask           -> f32 NDHWC   (shortcut, COUT=128)
//       2 = GN(groups of 4)+mask + add aux -> f32 NCDHW   (final,    COUT=128)
// ---------------------------------------------------------------------------
template<int TYPE, int COUT, int EPI>
__global__ void k_conv_wmma(const _Float16* __restrict__ in,
                            const _Float16* __restrict__ Wp,
                            const float* __restrict__ gamma,
                            const float* __restrict__ beta,
                            const float* __restrict__ maskf,
                            const float* __restrict__ aux,
                            _Float16* __restrict__ out_h,
                            float* __restrict__ out_f) {
  constexpr int MT   = COUT / 16;
  constexpr int NTHR = MT * 32;
  constexpr int SD   = (TYPE == 1) ? 1 : 3;         // halo span D
  constexpr int SH   = (TYPE == 2) ? 1 : 3;         // halo span H
  constexpr int SW   = (TYPE == 0) ? 16 : 18;       // halo span W
  constexpr int DMIN = (TYPE == 1) ? 0 : -1;
  constexpr int HMIN = (TYPE == 2) ? 0 : -1;
  constexpr int WMIN = (TYPE == 0) ? 0 : -1;
  constexpr int NST  = SD * SH * SW;                // staged voxels (<=144 -> 18KB)

  __shared__ alignas(64) _Float16 smem[NST * CIN];

  int tid  = threadIdx.x;
  int lane = tid & 31;
  int wv   = tid >> 5;
  int tile = blockIdx.x;
  int d  = tile / (HDIM * (WDIM / 16));
  int rr = tile % (HDIM * (WDIM / 16));
  int h  = rr / (WDIM / 16);
  int w0 = (rr % (WDIM / 16)) * 16;

  // ---- Stage halo block into LDS (zeros for out-of-range voxels) ----------
  {
    uint4* sm4 = (uint4*)smem;                       // 8 uint4 per voxel
    for (int i = tid; i < NST * 8; i += NTHR) {
      int vox = i >> 3, q = i & 7;
      int dz = vox / (SH * SW);
      int r2 = vox % (SH * SW);
      int hy = r2 / SW;
      int wx = r2 % SW;
      int sd = d + DMIN + dz, sh = h + HMIN + hy, sw = w0 + WMIN + wx;
      uint4 val = make_uint4(0u, 0u, 0u, 0u);
      if (((unsigned)sd < (unsigned)DDIM) && ((unsigned)sh < (unsigned)HDIM) &&
          ((unsigned)sw < (unsigned)WDIM)) {
        long src = (((long)sd * HDIM + sh) * WDIM + sw) * CIN + q * 8;
        val = *reinterpret_cast<const uint4*>(in + src);
      }
      sm4[i] = val;
    }
  }
  __syncthreads();

  // ---- WMMA main loop: ds_load B + global load A + v_wmma, EXEC all-ones ---------------
  int n     = lane & 15;                 // B/D column (voxel within tile)
  int khalf = (lane >> 4) << 4;          // K half held by this lane
  v8f acc = {};
  #pragma unroll
  for (int t = 0; t < 9; ++t) {
    int ta = t / 3, tb = t % 3;
    int od, oh, ow;
    if (TYPE == 0)      { od = ta - 1; oh = tb - 1; ow = 0;      }
    else if (TYPE == 1) { od = 0;      oh = ta - 1; ow = tb - 1; }
    else                { od = ta - 1; oh = 0;      ow = tb - 1; }
    int lvox = ((od - DMIN) * SH + (oh - HMIN)) * SW + (n + ow - WMIN);
    const _Float16* bp = smem + lvox * CIN + khalf;
    #pragma unroll
    for (int kc = 0; kc < 2; ++kc) {
      v16h bvec = *reinterpret_cast<const v16h*>(bp + kc * 32);
      v16h avec = *reinterpret_cast<const v16h*>(
          Wp + ((size_t)(((t * MT + wv) * 2 + kc) * 32 + lane) << 4));
      acc = __builtin_amdgcn_wmma_f32_16x16x32_f16(
          false, avec, false, bvec, (short)0, acc, false, false);
    }
  }

  // ---- Fused GroupNorm epilogue (groups are thread-local in D layout) -----
  long voxel = ((long)d * HDIM + h) * WDIM + w0 + n;
  int  mbase = wv * 16 + ((lane >> 4) << 3);   // first channel held by thread
  float m = maskf[voxel];

  if (EPI == 0) {
    // GN over channel pairs + LeakyReLU(0.01) + mask -> f16 NDHWC
    _Float16* o = out_h + voxel * 64 + mbase;
    #pragma unroll
    for (int g = 0; g < 4; ++g) {
      float a = acc[2 * g], b = acc[2 * g + 1];
      float mu = 0.5f * (a + b);
      float da = a - mu, db = b - mu;
      float inv = rsqrtf(0.5f * (da * da + db * db) + 1e-5f);
      int c0 = mbase + 2 * g;
      float ya = da * inv * gamma[c0]     + beta[c0];
      float yb = db * inv * gamma[c0 + 1] + beta[c0 + 1];
      ya = ya > 0.0f ? ya : 0.01f * ya;
      yb = yb > 0.0f ? yb : 0.01f * yb;
      o[2 * g]     = (_Float16)(ya * m);
      o[2 * g + 1] = (_Float16)(yb * m);
    }
  } else {
    // GN over groups of 4 channels
    #pragma unroll
    for (int g = 0; g < 2; ++g) {
      float tv[4], mu = 0.0f;
      #pragma unroll
      for (int j = 0; j < 4; ++j) { tv[j] = acc[4 * g + j]; mu += tv[j]; }
      mu *= 0.25f;
      float var = 0.0f;
      #pragma unroll
      for (int j = 0; j < 4; ++j) { tv[j] -= mu; var += tv[j] * tv[j]; }
      float inv = rsqrtf(var * 0.25f + 1e-5f);
      #pragma unroll
      for (int j = 0; j < 4; ++j) {
        int c = mbase + 4 * g + j;
        float y = (tv[j] * inv * gamma[c] + beta[c]) * m;
        if (EPI == 1) {
          out_f[voxel * 128 + c] = y;                       // shortcut, NDHWC
        } else {
          y += aux[voxel * 128 + c];                        // + shortcut
          out_f[(size_t)c * NVOX + voxel] = y;              // final, NCDHW
        }
      }
    }
  }
}

extern "C" void kernel_launch(void* const* d_in, const int* in_sizes, int n_in,
                              void* d_out, int out_size, void* d_ws, size_t ws_size,
                              hipStream_t stream) {
  (void)in_sizes; (void)n_in; (void)out_size; (void)ws_size;
  const float* x   = (const float*)d_in[0];
  // d_in[1] = mask (bool) -- re-derived from pre-masked x instead.
  const float* W1a = (const float*)d_in[2];
  const float* g1a = (const float*)d_in[3];
  const float* b1a = (const float*)d_in[4];
  const float* W1b = (const float*)d_in[5];
  const float* g1b = (const float*)d_in[6];
  const float* b1b = (const float*)d_in[7];
  const float* W2a = (const float*)d_in[8];
  const float* g2a = (const float*)d_in[9];
  const float* b2a = (const float*)d_in[10];
  const float* W2b = (const float*)d_in[11];
  const float* g2b = (const float*)d_in[12];
  const float* b2b = (const float*)d_in[13];

  char* ws = (char*)d_ws;
  _Float16* wp1a = (_Float16*)(ws + WP1A_OFF);
  _Float16* wp1b = (_Float16*)(ws + WP1B_OFF);
  _Float16* wp2a = (_Float16*)(ws + WP2A_OFF);
  _Float16* wp2b = (_Float16*)(ws + WP2B_OFF);
  float*    mk   = (float*)   (ws + MASK_OFF);
  _Float16* xt   = (_Float16*)(ws + XT_OFF);
  _Float16* s16  = (_Float16*)(ws + S16_OFF);
  float*    traw = (float*)   (ws + TRAW_OFF);
  float*    outp = (float*)d_out;

  // Weight repack (tiny)
  k_pack_weights<<<(9 * 64 * 64 + 255) / 256, 256, 0, stream>>>(W1a, wp1a, 64);
  k_pack_weights<<<(9 * 128 * 64 + 255) / 256, 256, 0, stream>>>(W1b, wp1b, 128);
  k_pack_weights<<<(9 * 64 * 64 + 255) / 256, 256, 0, stream>>>(W2a, wp2a, 64);
  k_pack_weights<<<(9 * 128 * 64 + 255) / 256, 256, 0, stream>>>(W2b, wp2b, 128);

  // Layout transform + mask derivation
  k_transform<<<NVOX / 256, 256, 0, stream>>>(x, xt, mk);

  const int tiles = DDIM * HDIM * (WDIM / 16);   // 16384 voxel tiles

  // Branch 1: conv(3,3,1)+GN+LReLU -> s16 ; conv(1,3,3)+GN -> shortcut (traw)
  k_conv_wmma<0, 64, 0><<<tiles, 128, 0, stream>>>(
      xt, wp1a, g1a, b1a, mk, nullptr, s16, nullptr);
  k_conv_wmma<1, 128, 1><<<tiles, 256, 0, stream>>>(
      s16, wp1b, g1b, b1b, mk, nullptr, nullptr, traw);

  // Branch 2: conv(3,1,3)+GN+LReLU -> s16 ; conv(3,3,1)+GN + add -> out NCDHW
  k_conv_wmma<2, 64, 0><<<tiles, 128, 0, stream>>>(
      xt, wp2a, g2a, b2a, mk, nullptr, s16, nullptr);
  k_conv_wmma<0, 128, 2><<<tiles, 256, 0, stream>>>(
      s16, wp2b, g2b, b2b, mk, traw, nullptr, outp);
}
